// FusionNeXtBackbone_69389491634575
// MI455X (gfx1250) — matrix operand order repaired and checked by reading, not compiled, not run on hardware
//
#include <hip/hip_runtime.h>
#include <hip/hip_bf16.h>

// ---------------------------------------------------------------------------
// Types & helpers
// ---------------------------------------------------------------------------
typedef __attribute__((ext_vector_type(16))) __bf16 v16bf;
typedef __attribute__((ext_vector_type(8)))  float  v8f;

#define DEV static __device__ __forceinline__

DEV unsigned short f2bf(float f) {
    unsigned u = __float_as_uint(f);
    unsigned r = u + 0x7fffu + ((u >> 16) & 1u);   // round-to-nearest-even
    return (unsigned short)(r >> 16);
}

union Frag {
    v16bf bf;
    uint4 q[2];
};

DEV v8f wmma_bf16(v16bf a, v16bf b, v8f c) {
    // (neg_a, A, neg_b, B, c_mod, C, reuse_a, reuse_b)
    return __builtin_amdgcn_wmma_f32_16x16x32_bf16(false, a, false, b, (short)0, c,
                                                   false, false);
}

// A fragment (16x32, M x K), src row-major bf16, leading dim ld.
// Layout used here (same as load_b): lane l -> row m0 + (l&15); K = 16*g + 0..15 contiguous
// (g = l>>4) -> 32B chunk.
DEV v16bf load_a(const unsigned short* src, int ld, int m0, int k0, int lane) {
    int m = lane & 15, g = lane >> 4;
    const unsigned short* p = src + (size_t)(m0 + m) * ld + k0 + g * 16;
    Frag f;
    f.q[0] = *(const uint4*)p;
    f.q[1] = *(const uint4*)(p + 8);
    return f.bf;
}

// B fragment (32x16, K x N) where B[k][n] = src[n0+n][k0+k] (src row-major [N,K]).
// Layout used here (same as load_a): lane l -> col n0 + (l&15); K = 16*g + 0..15 contiguous -> 32B chunk.
DEV v16bf load_b(const unsigned short* src, int ld, int n0, int k0, int lane) {
    int n = lane & 15, g = lane >> 4;
    const unsigned short* p = src + (size_t)(n0 + n) * ld + k0 + g * 16;
    Frag f;
    f.q[0] = *(const uint4*)p;
    f.q[1] = *(const uint4*)(p + 8);
    return f.bf;
}

// Register-blocked 32(M) x 64(N) wave tile: 2 A-frags x 4 B-frags -> 8 WMMA
// per 32-deep K step, 6 fragment loads (0.75 loads per WMMA).
// K loop unroll capped at 2 to keep live VGPRs < 256 (avoids s_set_vgpr_msb
// churn and keeps >=4 waves/SIMD for latency hiding).
struct Acc2x4 {
    v8f a[2][4];
};

DEV Acc2x4 gemm_tile_2x4(const unsigned short* __restrict__ A,
                         const unsigned short* __restrict__ W, int K,
                         int m0, int n0, int lane) {
    Acc2x4 acc;
    #pragma unroll
    for (int i = 0; i < 2; ++i)
        #pragma unroll
        for (int j = 0; j < 4; ++j) acc.a[i][j] = (v8f){};
    #pragma unroll 2
    for (int k0 = 0; k0 < K; k0 += 32) {
        v16bf a0 = load_a(A, K, m0, k0, lane);
        v16bf a1 = load_a(A, K, m0 + 16, k0, lane);
        #pragma unroll
        for (int j = 0; j < 4; ++j) {
            v16bf b = load_b(W, K, n0 + 16 * j, k0, lane);
            acc.a[0][j] = wmma_bf16(a0, b, acc.a[0][j]);
            acc.a[1][j] = wmma_bf16(a1, b, acc.a[1][j]);
        }
    }
    return acc;
}

DEV float block_sum(float v, float* red, int tid) {
    #pragma unroll
    for (int o = 16; o; o >>= 1) v += __shfl_xor(v, o, 32);
    if ((tid & 31) == 0) red[tid >> 5] = v;
    __syncthreads();
    float r = 0.f;
    #pragma unroll
    for (int i = 0; i < 8; ++i) r += red[i];
    __syncthreads();
    return r;
}

// ---------------------------------------------------------------------------
// Constants (problem sizes)
// ---------------------------------------------------------------------------
#define BB 4
#define SS 4096
#define CC 512
#define HH 8
#define DD 64
#define RR 128
#define W3 384          // 3R window
#define NBLK 32
#define MTOK (BB * SS)  // 16384
#define HID 2048

// GEMM wave-tile mapping: block = 8 waves arranged 2(M) x 4(N);
// block tile = 64(M) x 256(N); wave tile = 32 x 64.
DEV void gemm_coords(int* m0, int* n0) {
    int wave = threadIdx.x >> 5;
    *m0 = blockIdx.x * 64 + (wave & 1) * 32;
    *n0 = blockIdx.y * 256 + (wave >> 1) * 64;
}

// ---------------------------------------------------------------------------
// Kernel: f32 -> bf16 convert (weights)
// ---------------------------------------------------------------------------
__global__ void cvt_bf16_kernel(const float* __restrict__ src,
                                unsigned short* __restrict__ dst, int n) {
    int i = blockIdx.x * 256 + threadIdx.x;
    if (i < n) dst[i] = f2bf(src[i]);
}

// ---------------------------------------------------------------------------
// Kernel: LayerNorm over C=512, one token per block (256 thr, 2 elems/thr),
// writes bf16.
// ---------------------------------------------------------------------------
__global__ __launch_bounds__(256) void ln_bf16_kernel(
    const float* __restrict__ x, const float* __restrict__ gamma,
    const float* __restrict__ beta, unsigned short* __restrict__ out) {
    __shared__ float red[8];
    int t = blockIdx.x;
    int tid = threadIdx.x;
    const float* row = x + (size_t)t * CC;
    float a = row[tid], c = row[tid + 256];

    float sum = block_sum(a + c, red, tid);
    float mean = sum * (1.f / (float)CC);
    float da = a - mean, dc = c - mean;
    float var = block_sum(da * da + dc * dc, red, tid) * (1.f / (float)CC);
    float rstd = rsqrtf(var + 1e-5f);

    unsigned short* o = out + (size_t)t * CC;
    o[tid]       = f2bf(da * rstd * gamma[tid] + beta[tid]);
    o[tid + 256] = f2bf(dc * rstd * gamma[tid + 256] + beta[tid + 256]);
}

// ---------------------------------------------------------------------------
// Kernel: QKV GEMM [16384,512] x [1536,512]^T, scatter to Q/K/V [B,H,S,D] bf16
// grid (M/64, 1536/256), block 256
// ---------------------------------------------------------------------------
__global__ __launch_bounds__(256) void qkv_gemm_kernel(
    const unsigned short* __restrict__ xn, const unsigned short* __restrict__ W,
    const float* __restrict__ bias, unsigned short* __restrict__ Q,
    unsigned short* __restrict__ K, unsigned short* __restrict__ V) {
    int lane = threadIdx.x & 31;
    int m0, n0;
    gemm_coords(&m0, &n0);
    Acc2x4 acc = gemm_tile_2x4(xn, W, CC, m0, n0, lane);

    int nn = lane & 15, g = lane >> 4;
    #pragma unroll
    for (int j = 0; j < 4; ++j) {
        int n = n0 + j * 16 + nn;
        int mat = n >> 9;             // feature / 512 -> 0:q 1:k 2:v
        int h = (n >> 6) & 7;
        int d = n & 63;
        unsigned short* dst = (mat == 0) ? Q : ((mat == 1) ? K : V);
        float bv = bias[n];
        #pragma unroll
        for (int mi = 0; mi < 2; ++mi) {
            #pragma unroll
            for (int r = 0; r < 8; ++r) {
                int m = m0 + mi * 16 + g * 8 + r;   // token index
                int b = m >> 12;
                int s = m & (SS - 1);
                dst[(((size_t)(b * HH + h)) * SS + s) * DD + d] =
                    f2bf(acc.a[mi][j][r] + bv);
            }
        }
    }
}

// ---------------------------------------------------------------------------
// Kernel: sliding-window attention.
// grid (NB, B*H), block 256. Dynamic LDS:
//   sc  : float  [128][384]   (scores, reused in-place as bf16 probabilities)
//   vt  : ushort [64][384]    (V window transposed: d-major)
//   kval: uchar  [384]
// ---------------------------------------------------------------------------
__global__ __launch_bounds__(256) void attn_kernel(
    const unsigned short* __restrict__ Q, const unsigned short* __restrict__ K,
    const unsigned short* __restrict__ V, const unsigned char* __restrict__ pad,
    unsigned short* __restrict__ ctx) {
    extern __shared__ char smem[];
    float* sc = (float*)smem;                                   // 196608 B
    unsigned short* vt = (unsigned short*)(smem + RR * W3 * 4); // 49152 B
    unsigned char* kval = (unsigned char*)(smem + RR * W3 * 4 + DD * W3 * 2);

    int nb = blockIdx.x;
    int bh = blockIdx.y;                  // b*H + h
    int b = bh >> 3;
    int h = bh & 7;
    int qs0 = nb * RR;                    // first query of block
    const unsigned short* Qbh = Q + (size_t)bh * SS * DD;
    const unsigned short* Kbh = K + (size_t)bh * SS * DD;
    const unsigned short* Vbh = V + (size_t)bh * SS * DD;

    int tid = threadIdx.x;
    // stage key-validity flags and transposed V window
    for (int j = tid; j < W3; j += 256) {
        int s = qs0 - RR + j;
        kval[j] = (s >= 0 && s < SS && !pad[b * SS + s]) ? 1 : 0;
    }
    for (int idx = tid; idx < W3 * DD; idx += 256) {
        int j = idx >> 6, d = idx & 63;
        int s = qs0 - RR + j;
        vt[d * W3 + j] = (s >= 0 && s < SS) ? Vbh[(size_t)s * DD + d]
                                            : (unsigned short)0;
    }
    __syncthreads();

    int lane = tid & 31, wave = tid >> 5;
    int nn = lane & 15, g = lane >> 4;

    // ---- scores: S = Q Kt * scale, masked, into LDS ----
    v16bf qa0 = load_a(Qbh, DD, qs0 + wave * 16, 0, lane);
    v16bf qa1 = load_a(Qbh, DD, qs0 + wave * 16, 32, lane);
    const float scale = 0.125f;   // D^-1/2

    #pragma unroll 2
    for (int j = 0; j < 24; ++j) {
        int jj = j * 16 + nn;                 // window key index of this lane
        int skey = qs0 - RR + jj;
        int sclamp = min(max(skey, 0), SS - 1);
        const unsigned short* p = Kbh + (size_t)sclamp * DD + g * 16;
        Frag fb0, fb1;
        fb0.q[0] = *(const uint4*)p;          // K = d 0..31
        fb0.q[1] = *(const uint4*)(p + 8);
        fb1.q[0] = *(const uint4*)(p + 32);   // K = d 32..63
        fb1.q[1] = *(const uint4*)(p + 40);
        v8f acc = {};
        acc = wmma_bf16(qa0, fb0.bf, acc);
        acc = wmma_bf16(qa1, fb1.bf, acc);

        int kv = kval[jj];
        int kj = jj - RR;                     // key offset rel. to block start
        #pragma unroll
        for (int r = 0; r < 8; ++r) {
            int qi = wave * 16 + g * 8 + r;   // local query row 0..127
            int dlt = kj - qi;
            bool band = (dlt <= RR) && (dlt >= -RR);
            sc[qi * W3 + jj] = (band && kv) ? acc[r] * scale : -1e30f;
        }
    }
    __syncthreads();

    // ---- row softmax (fp32), repack probabilities in place as bf16 ----
    if (tid < RR) {
        float* row = sc + tid * W3;
        float mx = -1e30f;
        for (int j = 0; j < W3; ++j) mx = fmaxf(mx, row[j]);
        float sum = 0.f;
        for (int j = 0; j < W3; ++j) {
            float e = __expf(row[j] - mx);
            row[j] = e;
            sum += e;
        }
        float inv = 1.f / sum;
        unsigned short* pr = (unsigned short*)row;  // bf16 write never passes f32 read
        for (int j = 0; j < W3; ++j) pr[j] = f2bf(row[j] * inv);
    }
    __syncthreads();

    // ---- ctx = P (128x384) @ Vw (384x64), A-frag reused across 4 d-tiles ----
    v8f cacc[4];
    #pragma unroll
    for (int t = 0; t < 4; ++t) cacc[t] = (v8f){};
    const unsigned short* prow = (const unsigned short*)(sc + (wave * 16 + nn) * W3);
    #pragma unroll 2
    for (int ks = 0; ks < 12; ++ks) {
        int k0 = ks * 32;
        Frag fa;
        fa.q[0] = *(const uint4*)(prow + k0 + g * 16);
        fa.q[1] = *(const uint4*)(prow + k0 + g * 16 + 8);
        #pragma unroll
        for (int t = 0; t < 4; ++t) {
            const unsigned short* pv = vt + (t * 16 + nn) * W3 + k0 + g * 16;
            Frag fv;
            fv.q[0] = *(const uint4*)pv;
            fv.q[1] = *(const uint4*)(pv + 8);
            cacc[t] = wmma_bf16(fa.bf, fv.bf, cacc[t]);
        }
    }
    #pragma unroll
    for (int t = 0; t < 4; ++t) {
        int d = t * 16 + nn;
        #pragma unroll
        for (int r = 0; r < 8; ++r) {
            int qi = wave * 16 + g * 8 + r;
            int s = qs0 + qi;
            bool qv = !pad[b * SS + s];
            ctx[((size_t)(b * SS + s)) * CC + h * DD + d] =
                f2bf(qv ? cacc[t][r] : 0.f);
        }
    }
}

// ---------------------------------------------------------------------------
// Kernel: proj GEMM + bias + x residual + padding zero -> h (f32)
// ---------------------------------------------------------------------------
__global__ __launch_bounds__(256) void proj_gemm_kernel(
    const unsigned short* __restrict__ ctxb, const unsigned short* __restrict__ W,
    const float* __restrict__ bias, const float* __restrict__ x,
    const unsigned char* __restrict__ pad, float* __restrict__ hbuf) {
    int lane = threadIdx.x & 31;
    int m0, n0;
    gemm_coords(&m0, &n0);
    Acc2x4 acc = gemm_tile_2x4(ctxb, W, CC, m0, n0, lane);
    int nn = lane & 15, g = lane >> 4;
    #pragma unroll
    for (int j = 0; j < 4; ++j) {
        int n = n0 + j * 16 + nn;
        float bv = bias[n];
        #pragma unroll
        for (int mi = 0; mi < 2; ++mi) {
            #pragma unroll
            for (int r = 0; r < 8; ++r) {
                int m = m0 + mi * 16 + g * 8 + r;
                float v = acc.a[mi][j][r] + bv + x[(size_t)m * CC + n];
                hbuf[(size_t)m * CC + n] = pad[m] ? 0.f : v;
            }
        }
    }
}

// ---------------------------------------------------------------------------
// Kernel: MLP1 GEMM + bias + exact GELU -> act bf16 [16384,2048]
// ---------------------------------------------------------------------------
__global__ __launch_bounds__(256) void mlp1_gemm_kernel(
    const unsigned short* __restrict__ hn, const unsigned short* __restrict__ W1,
    const float* __restrict__ b1, unsigned short* __restrict__ act) {
    int lane = threadIdx.x & 31;
    int m0, n0;
    gemm_coords(&m0, &n0);
    Acc2x4 acc = gemm_tile_2x4(hn, W1, CC, m0, n0, lane);
    int nn = lane & 15, g = lane >> 4;
    #pragma unroll
    for (int j = 0; j < 4; ++j) {
        int n = n0 + j * 16 + nn;
        float bv = b1[n];
        #pragma unroll
        for (int mi = 0; mi < 2; ++mi) {
            #pragma unroll
            for (int r = 0; r < 8; ++r) {
                int m = m0 + mi * 16 + g * 8 + r;
                float v = acc.a[mi][j][r] + bv;
                float gel = 0.5f * v * (1.f + erff(v * 0.70710678f));
                act[(size_t)m * HID + n] = f2bf(gel);
            }
        }
    }
}

// ---------------------------------------------------------------------------
// Kernel: MLP2 GEMM + bias + h residual + padding zero -> out f32
// ---------------------------------------------------------------------------
__global__ __launch_bounds__(256) void mlp2_gemm_kernel(
    const unsigned short* __restrict__ act, const unsigned short* __restrict__ W2,
    const float* __restrict__ b2, const float* __restrict__ hbuf,
    const unsigned char* __restrict__ pad, float* __restrict__ out) {
    int lane = threadIdx.x & 31;
    int m0, n0;
    gemm_coords(&m0, &n0);
    Acc2x4 acc = gemm_tile_2x4(act, W2, HID, m0, n0, lane);
    int nn = lane & 15, g = lane >> 4;
    #pragma unroll
    for (int j = 0; j < 4; ++j) {
        int n = n0 + j * 16 + nn;
        float bv = b2[n];
        #pragma unroll
        for (int mi = 0; mi < 2; ++mi) {
            #pragma unroll
            for (int r = 0; r < 8; ++r) {
                int m = m0 + mi * 16 + g * 8 + r;
                float v = acc.a[mi][j][r] + bv + hbuf[(size_t)m * CC + n];
                out[(size_t)m * CC + n] = pad[m] ? 0.f : v;
            }
        }
    }
}

// ---------------------------------------------------------------------------
// Launch
// ---------------------------------------------------------------------------
extern "C" void kernel_launch(void* const* d_in, const int* in_sizes, int n_in,
                              void* d_out, int out_size, void* d_ws, size_t ws_size,
                              hipStream_t stream) {
    const float* x             = (const float*)d_in[0];
    const unsigned char* pad   = (const unsigned char*)d_in[1];  // bool mask
    const float* gamma1        = (const float*)d_in[2];
    const float* beta1         = (const float*)d_in[3];
    const float* Wqkv          = (const float*)d_in[4];
    const float* bqkv          = (const float*)d_in[5];
    const float* Wproj         = (const float*)d_in[6];
    const float* bproj         = (const float*)d_in[7];
    const float* gamma2        = (const float*)d_in[8];
    const float* beta2         = (const float*)d_in[9];
    const float* W1            = (const float*)d_in[10];
    const float* b1            = (const float*)d_in[11];
    const float* W2            = (const float*)d_in[12];
    const float* b2            = (const float*)d_in[13];
    float* out                 = (float*)d_out;

    char* ws = (char*)d_ws;
    size_t off = 0;
    auto alloc = [&](size_t bytes) {
        void* p = ws + off;
        off += (bytes + 255) & ~(size_t)255;
        return p;
    };

    unsigned short* wqkv_bf  = (unsigned short*)alloc((size_t)3 * CC * CC * 2);
    unsigned short* wproj_bf = (unsigned short*)alloc((size_t)CC * CC * 2);
    unsigned short* w1_bf    = (unsigned short*)alloc((size_t)HID * CC * 2);
    unsigned short* w2_bf    = (unsigned short*)alloc((size_t)CC * HID * 2);
    unsigned short* xn_bf    = (unsigned short*)alloc((size_t)MTOK * CC * 2);
    unsigned short* Qb       = (unsigned short*)alloc((size_t)MTOK * CC * 2);
    unsigned short* Kb       = (unsigned short*)alloc((size_t)MTOK * CC * 2);
    unsigned short* Vb       = (unsigned short*)alloc((size_t)MTOK * CC * 2);
    unsigned short* ctxb     = (unsigned short*)alloc((size_t)MTOK * CC * 2);
    float*          hbuf     = (float*)alloc((size_t)MTOK * CC * 4);
    unsigned short* hn_bf    = (unsigned short*)alloc((size_t)MTOK * CC * 2);
    unsigned short* act_bf   = (unsigned short*)alloc((size_t)MTOK * HID * 2);

    // 0) weight conversions
    {
        int n;
        n = 3 * CC * CC;
        cvt_bf16_kernel<<<(n + 255) / 256, 256, 0, stream>>>(Wqkv, wqkv_bf, n);
        n = CC * CC;
        cvt_bf16_kernel<<<(n + 255) / 256, 256, 0, stream>>>(Wproj, wproj_bf, n);
        n = HID * CC;
        cvt_bf16_kernel<<<(n + 255) / 256, 256, 0, stream>>>(W1, w1_bf, n);
        n = CC * HID;
        cvt_bf16_kernel<<<(n + 255) / 256, 256, 0, stream>>>(W2, w2_bf, n);
    }

    // 1) LN1 -> xn (bf16)
    ln_bf16_kernel<<<MTOK, 256, 0, stream>>>(x, gamma1, beta1, xn_bf);

    // 2) QKV GEMM -> Q/K/V [B,H,S,D] bf16  (block tile 64M x 256N)
    qkv_gemm_kernel<<<dim3(MTOK / 64, (3 * CC) / 256), 256, 0, stream>>>(
        xn_bf, wqkv_bf, bqkv, Qb, Kb, Vb);

    // 3) sliding-window attention -> ctx bf16 [B,S,C]
    {
        size_t smem = (size_t)RR * W3 * 4 + (size_t)DD * W3 * 2 + W3;
        attn_kernel<<<dim3(NBLK, BB * HH), 256, smem, stream>>>(Qb, Kb, Vb, pad, ctxb);
    }

    // 4) proj GEMM + residual + mask -> h (f32)
    proj_gemm_kernel<<<dim3(MTOK / 64, CC / 256), 256, 0, stream>>>(
        ctxb, wproj_bf, bproj, x, pad, hbuf);

    // 5) LN2 -> hn (bf16)
    ln_bf16_kernel<<<MTOK, 256, 0, stream>>>(hbuf, gamma2, beta2, hn_bf);

    // 6) MLP1 + GELU -> act (bf16)
    mlp1_gemm_kernel<<<dim3(MTOK / 64, HID / 256), 256, 0, stream>>>(
        hn_bf, w1_bf, b1, act_bf);

    // 7) MLP2 + residual + mask -> out (f32)
    mlp2_gemm_kernel<<<dim3(MTOK / 64, CC / 256), 256, 0, stream>>>(
        act_bf, w2_bf, b2, hbuf, pad, out);

    (void)in_sizes; (void)n_in; (void)out_size; (void)ws_size;
}
